// LSTM_300647711208
// MI455X (gfx1250) — compile-verified, matrix operand order checked by reading, not run on hardware
//
#include <hip/hip_runtime.h>
#include <hip/hip_bf16.h>

typedef __attribute__((ext_vector_type(16))) _Float16 v16h;
typedef __attribute__((ext_vector_type(8)))  float    v8f;

#define SEQ_LEN 2048
#define BATCH   512
#define HIDDEN  25

__device__ __forceinline__ float sigf(float x) {
    float e = __expf(-x);
    return __builtin_amdgcn_rcpf(1.0f + e);
}
__device__ __forceinline__ float tanhf_fast(float x) {
    // tanh(x) = 2*sigmoid(2x) - 1
    float e = __expf(-2.0f * x);
    return __builtin_amdgcn_rcpf(1.0f + e) * 2.0f - 1.0f;
}

// One wave32 per block; each block owns a 16-row batch tile and walks the
// full T=2048 recurrence with h held as a WMMA A-matrix in VGPRs.
__global__ void __launch_bounds__(32, 1)
lstm_wmma_kernel(const float* __restrict__ x,      // [T, B]
                 const float* __restrict__ Wih,    // [100, 1]
                 const float* __restrict__ Whh,    // [100, 25]
                 const float* __restrict__ bih,    // [100]
                 const float* __restrict__ bhh,    // [100]
                 const float* __restrict__ Wlin,   // [1, 25]
                 const float* __restrict__ blin_p, // [1]
                 float* __restrict__ out)          // [T, B]
{
    const int lane  = threadIdx.x;        // 0..31
    const int lo    = lane & 15;          // column-within-tile / M index
    const int hi    = lane >> 4;          // lane half
    const int bbase = blockIdx.x * 16;    // batch tile origin

    // LDS transpose buffer for h: [m][k] row-major, 16x32 f16, 1KB.
    __shared__ alignas(16) _Float16 hbuf[16 * 32];

    // ---------------- one-time setup: weights into VGPRs -----------------
    // B-matrix layout assumption (32x16, f16): lane holds column n = lane%16;
    // VGPRs 0..3 hold k = (0..7) + hi*8, VGPRs 4..7 hold k = (16..23) + hi*8,
    // two f16 per VGPR (k, k+1).  Gate padding: gate G occupies padded cols
    // [32G, 32G+32); valid cols are j<25 -> original col n = 25*G + j.
    v16h Bt[8];
    float bl[8], wl[8];
#pragma unroll
    for (int t8 = 0; t8 < 8; ++t8) {
        const int G = t8 >> 1;
        const int j = ((t8 & 1) << 4) + lo;    // 0..31 within gate block
        const int n = 25 * G + j;              // original gate column
        const bool jvalid = (j < HIDDEN);
        v16h b;
#pragma unroll
        for (int e = 0; e < 16; ++e) {
            const int v = e >> 1, d = e & 1;
            const int k = ((v < 4) ? (2 * v) : (16 + 2 * (v - 4))) + hi * 8 + d;
            float w = (jvalid && k < HIDDEN) ? Whh[n * HIDDEN + k] : 0.0f;
            b[e] = (_Float16)w;
        }
        Bt[t8] = b;
        bl[t8] = jvalid ? (bih[n] + bhh[n]) : 0.0f;
        wl[t8] = jvalid ? Wih[n] : 0.0f;
    }

    // Output-projection B: only column n==0 nonzero (= W_lin[k]).
    v16h Blin;
#pragma unroll
    for (int e = 0; e < 16; ++e) {
        const int v = e >> 1, d = e & 1;
        const int k = ((v < 4) ? (2 * v) : (16 + 2 * (v - 4))) + hi * 8 + d;
        float w = (lo == 0 && k < HIDDEN) ? Wlin[k] : 0.0f;
        Blin[e] = (_Float16)w;
    }
    const float blin = blin_p[0];

    // ---------------- state ----------------
    v16h hA;                                  // h_{t-1} in A-matrix layout
#pragma unroll
    for (int e = 0; e < 16; ++e) hA[e] = (_Float16)0.0f;
    float cst[16];                            // c state: [tt][r]
#pragma unroll
    for (int e = 0; e < 16; ++e) cst[e] = 0.0f;

    // Software-pipelined x loads: this lane needs rows m = r + 8*hi.
    float xcur[8], xnxt[8];
#pragma unroll
    for (int r = 0; r < 8; ++r)
        xcur[r] = x[0 * BATCH + bbase + r + 8 * hi];

    // ---------------- recurrence ----------------
    for (int t = 0; t < SEQ_LEN; ++t) {
        // (1) issue next step's x loads early (off the critical chain)
        const int tn = (t + 1 < SEQ_LEN) ? (t + 1) : t;
        __builtin_prefetch(&x[(size_t)tn * BATCH + bbase], 0, 3);
#pragma unroll
        for (int r = 0; r < 8; ++r)
            xnxt[r] = x[(size_t)tn * BATCH + bbase + r + 8 * hi];

        // (2) gate accumulators: C = bias + x_m * W_ih[n]  (D-layout)
        v8f acc[8];
#pragma unroll
        for (int t8 = 0; t8 < 8; ++t8) {
#pragma unroll
            for (int r = 0; r < 8; ++r)
                acc[t8][r] = bl[t8] + xcur[r] * wl[t8];
        }

        // (3) 8x v_wmma_f32_16x16x32_f16: gates += h_{t-1} @ W_hh^T
#pragma unroll
        for (int t8 = 0; t8 < 8; ++t8) {
            acc[t8] = __builtin_amdgcn_wmma_f32_16x16x32_f16(
                false, hA, false, Bt[t8], (short)0, acc[t8], false, false);
        }

        // (4) fused gate activations + state update; scatter h_t (f16) to LDS
        //     element (tt, r) -> m = r + 8*hi, j = 16*tt + lo
#pragma unroll
        for (int tt = 0; tt < 2; ++tt) {
#pragma unroll
            for (int r = 0; r < 8; ++r) {
                const float iv = sigf(acc[0 + tt][r]);
                const float fv = sigf(acc[2 + tt][r]);
                const float gv = tanhf_fast(acc[4 + tt][r]);
                const float ov = sigf(acc[6 + tt][r]);
                float c = fv * cst[tt * 8 + r] + iv * gv;
                cst[tt * 8 + r] = c;
                const float h = ov * tanhf_fast(c);
                hbuf[(r + 8 * hi) * 32 + (tt * 16 + lo)] = (_Float16)h;
            }
        }

        // (5) gather h_t back in A-matrix layout (16B+16B per lane)
        {
            union { v16h v; uint4 q[2]; } hu;
            const _Float16* p = &hbuf[lo * 32 + hi * 8];
            hu.q[0] = *reinterpret_cast<const uint4*>(p);
            hu.q[1] = *reinterpret_cast<const uint4*>(p + 16);
            hA = hu.v;
        }

        // (6) output projection: pred_t = h_t @ W_lin^T + b_lin (9th WMMA)
        v8f cp;
#pragma unroll
        for (int r = 0; r < 8; ++r) cp[r] = blin;
        v8f pred = __builtin_amdgcn_wmma_f32_16x16x32_f16(
            false, hA, false, Blin, (short)0, cp, false, false);
        if (lo == 0) {
#pragma unroll
            for (int r = 0; r < 8; ++r)
                out[(size_t)t * BATCH + bbase + r + 8 * hi] = pred[r];
        }

        // (7) rotate pipelined x
#pragma unroll
        for (int r = 0; r < 8; ++r) xcur[r] = xnxt[r];
    }
}

extern "C" void kernel_launch(void* const* d_in, const int* in_sizes, int n_in,
                              void* d_out, int out_size, void* d_ws, size_t ws_size,
                              hipStream_t stream) {
    (void)in_sizes; (void)n_in; (void)out_size; (void)d_ws; (void)ws_size;
    const float* x    = (const float*)d_in[0];
    const float* Wih  = (const float*)d_in[1];
    const float* Whh  = (const float*)d_in[2];
    const float* bih  = (const float*)d_in[3];
    const float* bhh  = (const float*)d_in[4];
    const float* Wlin = (const float*)d_in[5];
    const float* blin = (const float*)d_in[6];
    float* out = (float*)d_out;

    dim3 grid(BATCH / 16);   // 32 batch tiles
    dim3 block(32);          // one wave32 per tile
    lstm_wmma_kernel<<<grid, block, 0, stream>>>(x, Wih, Whh, bih, bhh, Wlin, blin, out);
}
